// NT_Xent_10986526343809
// MI455X (gfx1250) — matrix operand order repaired and checked by reading, not compiled, not run on hardware
//
#include <hip/hip_runtime.h>
#include <hip/hip_bf16.h>

typedef __attribute__((ext_vector_type(16))) _Float16 v16h;
typedef __attribute__((ext_vector_type(8)))  _Float16 v8h;
typedef __attribute__((ext_vector_type(8)))  float    v8f;
typedef __attribute__((ext_vector_type(4)))  int      v4i;

#define NROWS 16384
#define BHALF 8192
#define DDIM  256
#define CSPLIT 4
#define ROWS_PER_BLOCK 128   // 8 waves x 16 rows
#define TEMP_INV 10.0f
#define LOG2E 1.44269504088896340736f

#define TILE_COLS 32
#define LDSROW    264                    // 256 halves + 8 pad -> 528B row stride
#define LDS_TILE  (TILE_COLS * LDSROW)   // halves per buffer

// ---- CDNA5 async global->LDS copy (ASYNCcnt-tracked) ----------------------
__device__ __forceinline__ void async_copy_b128(_Float16* lds, const _Float16* g) {
#if __has_builtin(__builtin_amdgcn_global_load_async_to_lds_b128)
  typedef __attribute__((address_space(1))) v4i* gp_t;
  typedef __attribute__((address_space(3))) v4i* lp_t;
  __builtin_amdgcn_global_load_async_to_lds_b128((gp_t)g, (lp_t)lds, 0, 0);
#else
  unsigned l32 = (unsigned)(unsigned long long)(const void*)lds;
  unsigned long long g64 = (unsigned long long)(const void*)g;
  asm volatile("global_load_async_to_lds_b128 %0, %1, off"
               :: "v"(l32), "v"(g64) : "memory");
#endif
}

__device__ __forceinline__ void wait_async0() {
#if __has_builtin(__builtin_amdgcn_s_wait_asynccnt)
  __builtin_amdgcn_s_wait_asynccnt(0);
#else
  asm volatile("s_wait_asynccnt 0" ::: "memory");
#endif
}

// ---------------------------------------------------------------------------
// Phase 1: fp32 row L2-norm (eps-clamped), write normalized rows as f16,
// row-major [N][D]. One wave per row: 32 lanes x 8 elements.
// ---------------------------------------------------------------------------
__global__ __launch_bounds__(256) void nt_xent_normalize(
    const float* __restrict__ zi, const float* __restrict__ zj,
    _Float16* __restrict__ zn)
{
  const int lane = threadIdx.x & 31;
  const int wave = threadIdx.x >> 5;
  const int row  = blockIdx.x * 8 + wave;
  const float* __restrict__ src =
      (row < BHALF) ? (zi + (size_t)row * DDIM)
                    : (zj + (size_t)(row - BHALF) * DDIM);
  float4 v0 = ((const float4*)src)[lane * 2 + 0];
  float4 v1 = ((const float4*)src)[lane * 2 + 1];
  float ss = v0.x*v0.x + v0.y*v0.y + v0.z*v0.z + v0.w*v0.w
           + v1.x*v1.x + v1.y*v1.y + v1.z*v1.z + v1.w*v1.w;
  #pragma unroll
  for (int m = 16; m >= 1; m >>= 1) ss += __shfl_xor(ss, m, 32);
  const float inv = 1.0f / fmaxf(sqrtf(ss), 1e-8f);
  v8h h;
  h[0] = (_Float16)(v0.x * inv); h[1] = (_Float16)(v0.y * inv);
  h[2] = (_Float16)(v0.z * inv); h[3] = (_Float16)(v0.w * inv);
  h[4] = (_Float16)(v1.x * inv); h[5] = (_Float16)(v1.y * inv);
  h[6] = (_Float16)(v1.z * inv); h[7] = (_Float16)(v1.w * inv);
  ((v8h*)(zn + (size_t)row * DDIM))[lane] = h;
}

// ---------------------------------------------------------------------------
// Phase 2: fused Gram-matrix + logsumexp with fixed shift 10.
// B tiles (32 cols x 256 K = 16KB) staged once per block into LDS via
// async global->LDS copies, double-buffered; 8 waves consume via ds_load.
// Each wave holds its 16-row A fragments resident in VGPRs.
// ---------------------------------------------------------------------------
__global__ __launch_bounds__(256) void nt_xent_sim(
    const _Float16* __restrict__ zn,
    float* __restrict__ sPart, float* __restrict__ pPart)
{
  __shared__ _Float16 lbuf[2][LDS_TILE];

  const int lane   = threadIdx.x & 31;
  const int wave   = threadIdx.x >> 5;
  const int rowBlk = blockIdx.x / CSPLIT;
  const int seg    = blockIdx.x % CSPLIT;
  const int rowBase = rowBlk * ROWS_PER_BLOCK + wave * 16;
  const int colBeg  = seg * (NROWS / CSPLIT);
  const int nTiles  = (NROWS / CSPLIT) / TILE_COLS;

  // ---- resident A fragments: 16-bit A 16x32 layout ----
  const int am = lane & 15;
  const int ao = (lane >> 4) * 8;
  const _Float16* __restrict__ arow = zn + (size_t)(rowBase + am) * DDIM;
  v16h a[8];
  #pragma unroll
  for (int kk = 0; kk < 8; ++kk) {
    v8h lo = *(const v8h*)(arow + kk * 32 + ao);
    v8h hi = *(const v8h*)(arow + kk * 32 + 16 + ao);
    #pragma unroll
    for (int i = 0; i < 8; ++i) { a[kk][i] = lo[i]; a[kk][i + 8] = hi[i]; }
  }

  // B fragment lane mapping (16-bit B 32x16 layout)
  const int cn = lane & 15;
  const int bo = (lane >> 4) * 8;
  const int rowL = rowBase + ((lane >> 4) << 3);

  float s[8], p[8];
  #pragma unroll
  for (int r = 0; r < 8; ++r) { s[r] = 0.0f; p[r] = 0.0f; }

  // ---- async stage of one 32-col B tile: 1024 x 16B chunks, 4 per thread --
  auto stage = [&](int buf, int c0) {
    const _Float16* gsrc = zn + (size_t)c0 * DDIM;
    #pragma unroll
    for (int i = 0; i < 4; ++i) {
      const int chunk = i * 256 + (int)threadIdx.x;   // 0..1023
      const int r   = chunk >> 5;                     // tile row 0..31
      const int c16 = chunk & 31;                     // 16B chunk in row
      async_copy_b128(&lbuf[buf][r * LDSROW + c16 * 8],
                      gsrc + (size_t)r * DDIM + c16 * 8);
    }
  };

  stage(0, colBeg);
  int buf = 0;
  for (int it = 0; it < nTiles; ++it) {
    const int c0 = colBeg + it * TILE_COLS;
    wait_async0();          // own async stores to LDS complete
    __syncthreads();        // everyone's complete; prev buffer free to refill
    if (it + 1 < nTiles) stage(buf ^ 1, c0 + TILE_COLS);

    const _Float16* l0 = &lbuf[buf][cn * LDSROW + bo];
    const _Float16* l1 = &lbuf[buf][(cn + 16) * LDSROW + bo];
    v8f acc0 = {}; v8f acc1 = {};
    #pragma unroll
    for (int kk = 0; kk < 8; ++kk) {
      v8h t0 = *(const v8h*)(l0 + kk * 32);
      v8h t1 = *(const v8h*)(l0 + kk * 32 + 16);
      v8h u0 = *(const v8h*)(l1 + kk * 32);
      v8h u1 = *(const v8h*)(l1 + kk * 32 + 16);
      v16h b0, b1;
      #pragma unroll
      for (int i = 0; i < 8; ++i) {
        b0[i] = t0[i]; b0[i + 8] = t1[i];
        b1[i] = u0[i]; b1[i + 8] = u1[i];
      }
      acc0 = __builtin_amdgcn_wmma_f32_16x16x32_f16(
          false, a[kk], false, b0, (short)0, acc0, false, false);
      acc1 = __builtin_amdgcn_wmma_f32_16x16x32_f16(
          false, a[kk], false, b1, (short)0, acc1, false, false);
    }
    #pragma unroll
    for (int r = 0; r < 8; ++r) {
      const int row = rowL + r;
      {
        const float x = acc0[r] * TEMP_INV;
        const int col = c0 + cn;
        p[r] += (col == (row ^ BHALF)) ? x : 0.0f;
        const float e = __builtin_amdgcn_exp2f((x - TEMP_INV) * LOG2E);
        s[r] += (col == row) ? 0.0f : e;
      }
      {
        const float x = acc1[r] * TEMP_INV;
        const int col = c0 + 16 + cn;
        p[r] += (col == (row ^ BHALF)) ? x : 0.0f;
        const float e = __builtin_amdgcn_exp2f((x - TEMP_INV) * LOG2E);
        s[r] += (col == row) ? 0.0f : e;
      }
    }
    buf ^= 1;
  }

  // reduce the 16-lane column groups (masks 1..8 stay within each half-wave)
  #pragma unroll
  for (int m = 8; m >= 1; m >>= 1) {
    #pragma unroll
    for (int r = 0; r < 8; ++r) {
      s[r] += __shfl_xor(s[r], m, 32);
      p[r] += __shfl_xor(p[r], m, 32);
    }
  }
  if ((lane & 15) == 0) {
    #pragma unroll
    for (int r = 0; r < 8; ++r) {
      const int row = rowL + r;
      sPart[(size_t)row * CSPLIT + seg] = s[r];
      pPart[(size_t)row * CSPLIT + seg] = p[r];
    }
  }
}

// ---------------------------------------------------------------------------
// Phase 3: combine segment partials, lse = 10 + log(s), loss = sum(lse-pos)/N.
// Single block, fixed reduction order -> deterministic; initializes d_out.
// ---------------------------------------------------------------------------
__global__ __launch_bounds__(256) void nt_xent_finalize(
    const float* __restrict__ sPart, const float* __restrict__ pPart,
    float* __restrict__ out)
{
  __shared__ float red[256];
  float acc = 0.0f;
  for (int row = threadIdx.x; row < NROWS; row += 256) {
    float s = 0.0f, p = 0.0f;
    #pragma unroll
    for (int k = 0; k < CSPLIT; ++k) {
      s += sPart[(size_t)row * CSPLIT + k];
      p += pPart[(size_t)row * CSPLIT + k];
    }
    acc += (TEMP_INV + logf(s)) - p;
  }
  red[threadIdx.x] = acc;
  __syncthreads();
  #pragma unroll
  for (int off = 128; off >= 1; off >>= 1) {
    if (threadIdx.x < off) red[threadIdx.x] += red[threadIdx.x + off];
    __syncthreads();
  }
  if (threadIdx.x == 0) out[0] = red[0] / (float)NROWS;
}

extern "C" void kernel_launch(void* const* d_in, const int* in_sizes, int n_in,
                              void* d_out, int out_size, void* d_ws, size_t ws_size,
                              hipStream_t stream) {
  const float* zi = (const float*)d_in[0];
  const float* zj = (const float*)d_in[1];
  // d_in[2] (logit_scale_m) is unused by the reference.

  _Float16* zn   = (_Float16*)d_ws;                              // 8 MB
  float*    sP   = (float*)((char*)d_ws + (size_t)NROWS * DDIM * sizeof(_Float16));
  float*    pP   = sP + (size_t)NROWS * CSPLIT;                  // 2 x 256 KB

  nt_xent_normalize<<<NROWS / 8, 256, 0, stream>>>(zi, zj, zn);
  nt_xent_sim<<<(NROWS / ROWS_PER_BLOCK) * CSPLIT, 256, 0, stream>>>(zn, sP, pP);
  nt_xent_finalize<<<1, 256, 0, stream>>>(sP, pP, (float*)d_out);
}
